// StatEncoder_89154931130509
// MI455X (gfx1250) — matrix operand order repaired and checked by reading, not compiled, not run on hardware
//
#include <hip/hip_runtime.h>
#include <hip/hip_bf16.h>
#include <math.h>

#define Bsz  2048
#define Wlen 128
#define Fin  8
#define Hdim 256
#define HP   264   // padded LDS stride for 256-wide rows (bf16/f32)
#define XP   40    // padded LDS stride for x tile (32 padded cols)
#define CP   520   // padded LDS stride for concat [h_fwd | h_bwd] (512 cols)

typedef __attribute__((ext_vector_type(16))) __bf16 v16bf;
typedef __attribute__((ext_vector_type(8)))  float  v8f;

static __device__ __forceinline__ v16bf ld_frag(const __bf16* p0, const __bf16* p1) {
  union { v16bf v; uint4 q[2]; } u;
  u.q[0] = *(const uint4*)p0;
  u.q[1] = *(const uint4*)p1;
  return u.v;
}

static __device__ __forceinline__ v8f wmma_bf(v16bf a, v16bf b, v8f c) {
  return __builtin_amdgcn_wmma_f32_16x16x32_bf16(false, a, false, b, (short)0, c, false, false);
}

static __device__ __forceinline__ float sigm(float x) {
  return 1.0f / (1.0f + expf(-x));
}

// ---------------------------------------------------------------------------
// Prep: fp32 -> bf16 weight copies. Wih is zero-padded from K=8 to K=32.
// ---------------------------------------------------------------------------
__global__ __launch_bounds__(256) void prep_kernel(
    const float* __restrict__ Wih_f, const float* __restrict__ Whh_f,
    const float* __restrict__ Wp,
    __bf16* __restrict__ whh_bf, __bf16* __restrict__ wih_bf,
    __bf16* __restrict__ wp_bf) {
  int i = blockIdx.x * 256 + threadIdx.x;
  if (i < 3 * Hdim * Hdim)          // 768 x 256
    whh_bf[i] = (__bf16)Whh_f[i];
  if (i < 3 * Hdim * 32) {          // 768 x 32 (padded)
    int r = i >> 5, c = i & 31;
    wih_bf[i] = (__bf16)((c < Fin) ? Wih_f[r * Fin + c] : 0.0f);
  }
  if (i < Hdim * 2 * Hdim)          // 256 x 512
    wp_bf[i] = (__bf16)Wp[i];
}

// ---------------------------------------------------------------------------
// Kernel 1: forward GRU recurrence. One WG = 32 batch rows, 8 waves.
// Wave w owns hidden columns [32w, 32w+32): 2 N-tiles per gate.
// acc[mt][g*2+ntl] accumulates gi+gh for gates r,z and gh-only for gate n;
// accin[mt][ntl] holds the x-only contribution of the n gate.
// Weight base pointers are bounced through LDS every step so the compiler
// cannot hoist the (loop-invariant) fragment loads out of the t-loop and
// spill them to scratch; weights stream from L2 (384 KB, device-resident).
// ---------------------------------------------------------------------------
__global__ __launch_bounds__(256) void gru_fwd_kernel(
    const float* __restrict__ x,
    const float* __restrict__ bih, const float* __restrict__ bhh,
    const __bf16* __restrict__ whh_bf_in, const __bf16* __restrict__ wih_bf_in,
    __bf16* __restrict__ hfwd) {
  __shared__ __bf16 sh_h[32][HP];
  __shared__ __bf16 sh_x[32][XP];
  __shared__ const __bf16* sh_wb[2];   // opaque weight base pointers

  const int tid  = threadIdx.x;
  const int wv   = tid >> 5;
  const int lane = tid & 31;
  const int ln   = lane & 15;
  const int hi   = lane >> 4;          // 0 or 1
  const int kbA  = hi ? 8 : 0;         // A-fragment K base per lane half
  const int kbB  = hi ? 8 : 0;         // B-fragment K base per lane half
  const int row0 = blockIdx.x * 32;
  const int jbase = wv * 32;

  for (int i = tid; i < 32 * HP; i += 256) ((__bf16*)sh_h)[i] = (__bf16)0.0f;
  for (int i = tid; i < 32 * XP; i += 256) ((__bf16*)sh_x)[i] = (__bf16)0.0f;

  // Per-N-tile weight row offsets (loop invariant, cheap ints).
  int rowoffW[6], rowoffI[6];
#pragma unroll
  for (int nt = 0; nt < 6; ++nt) {
    int ntg = (nt >> 1) * 16 + wv * 2 + (nt & 1);       // global N-tile (0..47)
    rowoffW[nt] = (ntg * 16 + ln) * Hdim + kbB;
    rowoffI[nt] = (ntg * 16 + ln) * 32 + kbB;
  }

  // Loop-invariant biases for this lane's columns.
  float c_bih[2][3], c_bhh[2][3];
#pragma unroll
  for (int ntl = 0; ntl < 2; ++ntl) {
    int j = jbase + ntl * 16 + ln;
#pragma unroll
    for (int g = 0; g < 3; ++g) {
      c_bih[ntl][g] = bih[g * Hdim + j];
      c_bhh[ntl][g] = bhh[g * Hdim + j];
    }
  }

  float hreg[2][2][8];
#pragma unroll
  for (int a = 0; a < 2; ++a)
#pragma unroll
    for (int b = 0; b < 2; ++b)
#pragma unroll
      for (int r = 0; r < 8; ++r) hreg[a][b][r] = 0.0f;

  __syncthreads();

#pragma unroll 1
  for (int t = 0; t < Wlen; ++t) {
    { // stage x_t tile (cols 8..31 stay zero) + refresh opaque pointers
      int r = tid >> 3, c = tid & 7;
      sh_x[r][c] = (__bf16)x[(size_t)(row0 + r) * (Wlen * Fin) + t * Fin + c];
      if (tid == 0) { sh_wb[0] = whh_bf_in; sh_wb[1] = wih_bf_in; }
    }
    __syncthreads();
    const __bf16* wb  = sh_wb[0];   // unknown per iteration => loads stay in loop
    const __bf16* wib = sh_wb[1];

    v8f acc[2][6];
    v8f accin[2][2];
#pragma unroll
    for (int m = 0; m < 2; ++m) {
#pragma unroll
      for (int n = 0; n < 6; ++n) acc[m][n] = (v8f){0,0,0,0,0,0,0,0};
#pragma unroll
      for (int n = 0; n < 2; ++n) accin[m][n] = (v8f){0,0,0,0,0,0,0,0};
    }

    // h @ Whh.T : K = 256 in 8 chunks of 32
#pragma unroll 2
    for (int kc = 0; kc < 8; ++kc) {
      v16bf a0 = ld_frag(&sh_h[ln][kc * 32 + kbA],      &sh_h[ln][kc * 32 + kbA + 16]);
      v16bf a1 = ld_frag(&sh_h[16 + ln][kc * 32 + kbA], &sh_h[16 + ln][kc * 32 + kbA + 16]);
#pragma unroll
      for (int nt = 0; nt < 6; ++nt) {
        const __bf16* pb = wb + rowoffW[nt] + kc * 32;
        v16bf bf = ld_frag(pb, pb + 16);
        acc[0][nt] = wmma_bf(a0, bf, acc[0][nt]);
        acc[1][nt] = wmma_bf(a1, bf, acc[1][nt]);
      }
    }
    { // x_t @ Wih.T : one padded K-chunk. Gate n goes to separate accumulator.
      v16bf a0 = ld_frag(&sh_x[ln][kbA],      &sh_x[ln][kbA + 16]);
      v16bf a1 = ld_frag(&sh_x[16 + ln][kbA], &sh_x[16 + ln][kbA + 16]);
#pragma unroll
      for (int nt = 0; nt < 6; ++nt) {
        const __bf16* pb = wib + rowoffI[nt];
        v16bf bf = ld_frag(pb, pb + 16);
        if (nt < 4) {
          acc[0][nt] = wmma_bf(a0, bf, acc[0][nt]);
          acc[1][nt] = wmma_bf(a1, bf, acc[1][nt]);
        } else {
          accin[0][nt - 4] = wmma_bf(a0, bf, accin[0][nt - 4]);
          accin[1][nt - 4] = wmma_bf(a1, bf, accin[1][nt - 4]);
        }
      }
    }
    __syncthreads();

    // Gate update (wave-local on accumulator layout), fp32 h state in regs.
#pragma unroll
    for (int mt = 0; mt < 2; ++mt)
#pragma unroll
      for (int ntl = 0; ntl < 2; ++ntl)
#pragma unroll
        for (int r = 0; r < 8; ++r) {
          float pr  = acc[mt][ntl][r]     + c_bih[ntl][0] + c_bhh[ntl][0];
          float pz  = acc[mt][2 + ntl][r] + c_bih[ntl][1] + c_bhh[ntl][1];
          float ghn = acc[mt][4 + ntl][r] + c_bhh[ntl][2];
          float gin = accin[mt][ntl][r]   + c_bih[ntl][2];
          float rg = sigm(pr);
          float zg = sigm(pz);
          float ng = tanhf(gin + rg * ghn);
          float hv = (1.0f - zg) * ng + zg * hreg[mt][ntl][r];
          hreg[mt][ntl][r] = hv;
          int rw = mt * 16 + hi * 8 + r;
          sh_h[rw][jbase + ntl * 16 + ln] = (__bf16)hv;
        }
    // next iteration's barrier makes sh_h writes visible
  }

  // emit final hidden state as bf16 (consumed only by the projection GEMM)
#pragma unroll
  for (int mt = 0; mt < 2; ++mt)
#pragma unroll
    for (int ntl = 0; ntl < 2; ++ntl)
#pragma unroll
      for (int r = 0; r < 8; ++r) {
        int rw = mt * 16 + hi * 8 + r;
        hfwd[(size_t)(row0 + rw) * Hdim + jbase + ntl * 16 + ln] = (__bf16)hreg[mt][ntl][r];
      }
}

// ---------------------------------------------------------------------------
// Kernel 2: backward cell (h0=0 => gh=bhh_b), concat, projection WMMA,
// exact GeLU, LayerNorm, affine.
// ---------------------------------------------------------------------------
__global__ __launch_bounds__(256) void head_kernel(
    const float* __restrict__ x,
    const float* __restrict__ Wih_b, const float* __restrict__ bih_b,
    const float* __restrict__ bhh_b,
    const float* __restrict__ bp, const float* __restrict__ gamma,
    const float* __restrict__ beta,
    const __bf16* __restrict__ hfwd, const __bf16* __restrict__ wp_bf,
    float* __restrict__ out) {
  __shared__ __bf16 sh_cat[32][CP];
  __shared__ float  sh_y[32][HP];
  __shared__ float  sh_xl[32][8];
  __shared__ float  sh_mu[32];
  __shared__ float  sh_rs[32];

  const int tid  = threadIdx.x;
  const int wv   = tid >> 5;
  const int lane = tid & 31;
  const int ln   = lane & 15;
  const int hi   = lane >> 4;
  const int kbA  = hi ? 8 : 0;
  const int kbB  = hi ? 8 : 0;
  const int row0 = blockIdx.x * 32;

  // copy h_fwd tile into concat LDS (u32 granularity; strides are 4B aligned)
  for (int i = tid; i < 32 * 128; i += 256) {
    int r = i >> 7, c = i & 127;
    *(uint32_t*)&sh_cat[r][c * 2] =
        ((const uint32_t*)hfwd)[(size_t)(row0 + r) * 128 + c];
  }
  { // stage x[:, W-1, :]
    int r = tid >> 3, c = tid & 7;
    sh_xl[r][c] = x[(size_t)(row0 + r) * (Wlen * Fin) + (Wlen - 1) * Fin + c];
  }
  __syncthreads();

  { // backward GRU cell with h=0: thread t owns column jh=t for all 32 rows
    int jh = tid;
    float wr[8], wz[8], wn[8];
#pragma unroll
    for (int c = 0; c < 8; ++c) {
      wr[c] = Wih_b[jh * Fin + c];
      wz[c] = Wih_b[(Hdim + jh) * Fin + c];
      wn[c] = Wih_b[(2 * Hdim + jh) * Fin + c];
    }
    float br = bih_b[jh] + bhh_b[jh];
    float bz = bih_b[Hdim + jh] + bhh_b[Hdim + jh];
    float bni = bih_b[2 * Hdim + jh];
    float bnh = bhh_b[2 * Hdim + jh];
    for (int r = 0; r < 32; ++r) {
      float ir = br, iz = bz, inn = bni;
#pragma unroll
      for (int c = 0; c < 8; ++c) {
        float xv = sh_xl[r][c];
        ir += xv * wr[c]; iz += xv * wz[c]; inn += xv * wn[c];
      }
      float rg = sigm(ir), zg = sigm(iz);
      float ng = tanhf(inn + rg * bnh);
      sh_cat[r][Hdim + jh] = (__bf16)((1.0f - zg) * ng);
    }
  }
  __syncthreads();

  // [32,512] @ Wp.T[512,256] : wave w owns output N-tiles 2w, 2w+1
  v8f acc[2][2];
#pragma unroll
  for (int m = 0; m < 2; ++m)
#pragma unroll
    for (int n = 0; n < 2; ++n) acc[m][n] = (v8f){0,0,0,0,0,0,0,0};

#pragma unroll 2
  for (int kc = 0; kc < 16; ++kc) {
    v16bf a0 = ld_frag(&sh_cat[ln][kc * 32 + kbA],      &sh_cat[ln][kc * 32 + kbA + 16]);
    v16bf a1 = ld_frag(&sh_cat[16 + ln][kc * 32 + kbA], &sh_cat[16 + ln][kc * 32 + kbA + 16]);
#pragma unroll
    for (int ntl = 0; ntl < 2; ++ntl) {
      int n = (wv * 2 + ntl) * 16 + ln;
      const __bf16* pb = wp_bf + (size_t)n * (2 * Hdim) + kc * 32 + kbB;
      v16bf bf = ld_frag(pb, pb + 16);
      acc[0][ntl] = wmma_bf(a0, bf, acc[0][ntl]);
      acc[1][ntl] = wmma_bf(a1, bf, acc[1][ntl]);
    }
  }

  // bias + exact GeLU -> sh_y
#pragma unroll
  for (int mt = 0; mt < 2; ++mt)
#pragma unroll
    for (int ntl = 0; ntl < 2; ++ntl) {
      int col = (wv * 2 + ntl) * 16 + ln;
      float bpc = bp[col];
#pragma unroll
      for (int r = 0; r < 8; ++r) {
        float v = acc[mt][ntl][r] + bpc;
        float g = 0.5f * v * (1.0f + erff(v * 0.70710678118654752f));
        sh_y[mt * 16 + hi * 8 + r][col] = g;
      }
    }
  __syncthreads();

  if (tid < 32) {
    float s = 0.0f, s2 = 0.0f;
    for (int c = 0; c < Hdim; ++c) {
      float v = sh_y[tid][c];
      s += v; s2 += v * v;
    }
    float mu  = s * (1.0f / Hdim);
    float var = s2 * (1.0f / Hdim) - mu * mu;
    sh_mu[tid] = mu;
    sh_rs[tid] = rsqrtf(var + 1e-5f);
  }
  __syncthreads();

  for (int e = 0; e < 32; ++e) {
    int flat = tid * 32 + e;
    int r = flat >> 8, c = flat & 255;
    out[(size_t)(row0 + r) * Hdim + c] =
        (sh_y[r][c] - sh_mu[r]) * sh_rs[r] * gamma[c] + beta[c];
  }
}

// ---------------------------------------------------------------------------
// Launch
// ---------------------------------------------------------------------------
extern "C" void kernel_launch(void* const* d_in, const int* in_sizes, int n_in,
                              void* d_out, int out_size, void* d_ws, size_t ws_size,
                              hipStream_t stream) {
  const float* x     = (const float*)d_in[0];
  const float* Wih_f = (const float*)d_in[1];
  const float* Whh_f = (const float*)d_in[2];
  const float* bih_f = (const float*)d_in[3];
  const float* bhh_f = (const float*)d_in[4];
  const float* Wih_b = (const float*)d_in[5];
  // d_in[6] (Whh_b) unused: h0 == 0 so gh == bhh_b exactly.
  const float* bih_b = (const float*)d_in[7];
  const float* bhh_b = (const float*)d_in[8];
  const float* Wp    = (const float*)d_in[9];
  const float* bp    = (const float*)d_in[10];
  const float* gamma = (const float*)d_in[11];
  const float* beta  = (const float*)d_in[12];

  char* ws = (char*)d_ws;
  __bf16* whh_bf = (__bf16*)(ws + 0);         // 768*256*2 = 393216 B
  __bf16* wih_bf = (__bf16*)(ws + 393216);    // 768*32*2  =  49152 B
  __bf16* wp_bf  = (__bf16*)(ws + 442368);    // 256*512*2 = 262144 B
  __bf16* hfwd   = (__bf16*)(ws + 704512);    // 2048*256*2 = 1 MiB

  prep_kernel<<<768, 256, 0, stream>>>(Wih_f, Whh_f, Wp, whh_bf, wih_bf, wp_bf);
  gru_fwd_kernel<<<Bsz / 32, 256, 0, stream>>>(x, bih_f, bhh_f, whh_bf, wih_bf, hfwd);
  head_kernel<<<Bsz / 32, 256, 0, stream>>>(x, Wih_b, bih_b, bhh_b, bp, gamma, beta,
                                            hfwd, wp_bf, (float*)d_out);
}
